// MyModel_87522843559689
// MI455X (gfx1250) — matrix operand order repaired and checked by reading, not compiled, not run on hardware
//
#include <hip/hip_runtime.h>
#include <stdint.h>

typedef __attribute__((ext_vector_type(16))) __bf16 v16bf;
typedef __attribute__((ext_vector_type(8)))  float  v8f;

struct U32x8 { uint4 lo, hi; };   // 32 bytes -> one v16bf fragment

#define B_     100
#define T_     1403
#define V_     500
#define D_     100
#define U_     200
#define FOURU  800
#define KPAD_IN  128     // D padded to mult of 32
#define KPAD_REC 224     // U padded to mult of 32
#define NT     50        // 800 / 16 column tiles
#define MROWS  (B_*T_)   // 140300 flattened (b,t) rows
#define MT_A   ((MROWS + 15) / 16)   // 8769

// LDS layout for the scan kernel (bytes)
#define HBUF_BYTES (2 * 112 * KPAD_REC * 2)        // 100352
#define CBUF_OFF   HBUF_BYTES
#define CBUF_BYTES (112 * U_ * 4)                  // 89600
#define STAGE_OFF  (CBUF_OFF + CBUF_BYTES)         // 189952
#define STAGE_BYTES (32 * 256 * 4)                 // 32768
#define SMEM_BYTES (STAGE_OFF + STAGE_BYTES)       // 222720

__device__ __forceinline__ unsigned short f2bf(float x) {
  unsigned u = __builtin_bit_cast(unsigned, x);
  u += 0x7FFFu + ((u >> 16) & 1u);               // round-to-nearest-even
  return (unsigned short)(u >> 16);
}

#if __has_builtin(__builtin_amdgcn_tanhf)
__device__ __forceinline__ float fast_tanh(float x) { return __builtin_amdgcn_tanhf(x); }
#else
__device__ __forceinline__ float fast_tanh(float x) { return tanhf(x); }
#endif

__device__ __forceinline__ float fast_sigmoid(float x) {
  return 0.5f * fast_tanh(0.5f * x) + 0.5f;     // one v_tanh_f32
}

// ---------------------------------------------------------------------------
// Prep: bf16 embedding table (K-padded), permuted bias, and weight matrices
// pre-swizzled into WMMA B-fragment order.
// Gate permutation: col' = 4*j + g  <->  original col = g*200 + j.
// B-fragment layout (16-bit B, 32x16): lane L holds column N = nt*16 + (L&15),
// K = kc*32 + ((L&16)?16:0) + e  for e = 0..15 (contiguous in K).
// ---------------------------------------------------------------------------
__global__ void prep_kernel(const float* __restrict__ emb_table,
                            const float* __restrict__ in_kernel,
                            const float* __restrict__ rec_kernel,
                            const float* __restrict__ bias,
                            unsigned short* __restrict__ embbf,
                            float* __restrict__ biasp,
                            unsigned short* __restrict__ kfrag,
                            unsigned short* __restrict__ wfrag) {
  const int nEmb = V_ * KPAD_IN;        // 64000
  const int nKf  = NT * 4 * 32 * 16;    // 102400
  const int nWf  = NT * 7 * 32 * 16;    // 179200
  const int total = nEmb + FOURU + nKf + nWf;
  for (int idx = blockIdx.x * blockDim.x + threadIdx.x; idx < total;
       idx += gridDim.x * blockDim.x) {
    if (idx < nEmb) {
      int v = idx / KPAD_IN, k = idx % KPAD_IN;
      float x = (k < D_) ? emb_table[v * D_ + k] : 0.0f;
      embbf[idx] = f2bf(x);
    } else if (idx < nEmb + FOURU) {
      int c = idx - nEmb;
      biasp[c] = bias[(c & 3) * U_ + (c >> 2)];
    } else if (idx < nEmb + FOURU + nKf) {
      int i = idx - nEmb - FOURU;
      int e = i & 15, L = (i >> 4) & 31, r2 = i >> 9;
      int kc = r2 & 3, nt = r2 >> 2;
      int K = kc * 32 + ((L & 16) ? 16 : 0) + e;
      int N = nt * 16 + (L & 15);
      int oc = (N & 3) * U_ + (N >> 2);
      float x = (K < D_) ? in_kernel[K * FOURU + oc] : 0.0f;
      kfrag[i] = f2bf(x);
    } else {
      int i = idx - nEmb - FOURU - nKf;
      int e = i & 15, L = (i >> 4) & 31, r2 = i >> 9;
      int kc = r2 % 7, nt = r2 / 7;
      int K = kc * 32 + ((L & 16) ? 16 : 0) + e;
      int N = nt * 16 + (L & 15);
      int oc = (N & 3) * U_ + (N >> 2);
      float x = (K < U_) ? rec_kernel[K * FOURU + oc] : 0.0f;
      wfrag[i] = f2bf(x);
    }
  }
}

// ---------------------------------------------------------------------------
// Phase 1: xz[b*T+t, col'] = emb_table[tokens] @ kernel + bias  (WMMA bf16)
// One wave per 16x16 output tile, grid-stride over 8769*50 tiles.
// Only the very last m-tile is partial -> uniform branch, no per-element exec
// manipulation on the hot path.
// ---------------------------------------------------------------------------
__global__ void __launch_bounds__(256)
input_gemm_kernel(const int* __restrict__ tokens,
                  const unsigned short* __restrict__ embbf,
                  const unsigned short* __restrict__ kfrag,
                  const float* __restrict__ biasp,
                  float* __restrict__ xz) {
  const int lane = threadIdx.x & 31;
  const int gwave = blockIdx.x * (blockDim.x >> 5) + (threadIdx.x >> 5);
  const int nwaves = gridDim.x * (blockDim.x >> 5);
  const int lhi = (lane & 16) ? 1 : 0;
  const int l15 = lane & 15;
  const int nTiles = MT_A * NT;

  for (int tile = gwave; tile < nTiles; tile += nwaves) {
    const int mt = tile / NT;
    const int nt = tile % NT;
    int row  = mt * 16 + l15;
    int rowc = row < MROWS ? row : MROWS - 1;
    int tok  = tokens[rowc];
    const unsigned short* arow = embbf + (size_t)tok * KPAD_IN;
    float bv = biasp[nt * 16 + l15];
    v8f acc = {bv, bv, bv, bv, bv, bv, bv, bv};
    const int kb0 = lhi * 16;
#pragma unroll
    for (int kc = 0; kc < 4; ++kc) {
      const int kb = kb0 + kc * 32;
      U32x8 at, bt;
      at.lo = *(const uint4*)(arow + kb);
      at.hi = *(const uint4*)(arow + kb + 8);
      const unsigned short* bp = kfrag + (((size_t)nt * 4 + kc) * 32 + lane) * 16;
      bt.lo = *(const uint4*)(bp);
      bt.hi = *(const uint4*)(bp + 8);
      v16bf av = __builtin_bit_cast(v16bf, at);
      v16bf bvv = __builtin_bit_cast(v16bf, bt);
      acc = __builtin_amdgcn_wmma_f32_16x16x32_bf16(false, av, false, bvv,
                                                    (short)0, acc, false, false);
    }
    const int colg = nt * 16 + l15;
    const size_t base = (size_t)(mt * 16 + lhi * 8) * FOURU + colg;
    if (mt != MT_A - 1) {               // full tile: unconditional stores
#pragma unroll
      for (int r = 0; r < 8; ++r)
        xz[base + (size_t)r * FOURU] = acc[r];
    } else {                            // single partial tile
#pragma unroll
      for (int r = 0; r < 8; ++r) {
        int m = mt * 16 + r + lhi * 8;
        if (m < MROWS) xz[(size_t)m * FOURU + colg] = acc[r];
      }
    }
  }
}

// ---------------------------------------------------------------------------
// Gate math for one 16x16 z-tile (16 batch rows x 4 hidden units x 4 gates).
// GUARD=false for tiles whose 16 rows are all valid (mt < 6).
// ---------------------------------------------------------------------------
template <bool GUARD>
__device__ __forceinline__ void gate_update(const float* __restrict__ st,
                                            float* __restrict__ cbuf,
                                            unsigned short* __restrict__ hw,
                                            float* __restrict__ out,
                                            int mt, int nt, int t, int lane) {
#pragma unroll
  for (int q = 0; q < 2; ++q) {
    int p = lane + q * 32;
    int mloc = p & 15, jl = p >> 4;
    int m = mt * 16 + mloc;
    int j = nt * 4 + jl;
    float zi = st[mloc * 16 + 4 * jl + 0];
    float zf = st[mloc * 16 + 4 * jl + 1];
    float zg = st[mloc * 16 + 4 * jl + 2];
    float zo = st[mloc * 16 + 4 * jl + 3];
    float ig = fast_sigmoid(zi);
    float fg = fast_sigmoid(zf);
    float gg = fast_tanh(zg);
    float og = fast_sigmoid(zo);
    float cp = cbuf[m * U_ + j];
    float cn = fg * cp + ig * gg;
    float hn = og * fast_tanh(cn);
    cbuf[m * U_ + j] = cn;                 // static owner per (m,j)
    hw[m * KPAD_REC + j] = f2bf(hn);       // next step's A matrix
    if (!GUARD || m < B_) {
      size_t ob = ((size_t)m * T_ + t) * (2 * U_);
      out[ob + j] = hn;
      out[ob + U_ + j] = cn;
    }
  }
}

// ---------------------------------------------------------------------------
// Phase 2: persistent single-workgroup LSTM scan. 1024 threads = 32 waves on
// one WGP. h (bf16, double-buffered) and c (fp32) live in LDS (CDNA5 320KB).
// Each step: 350 16x16 tiles of z = xz[:,t,:] + h @ Wrec (7 bf16 WMMAs/tile),
// v_tanh_f32 gate math, prefetch of the next timestep's xz rows, and one
// workgroup barrier per step.
// ---------------------------------------------------------------------------
__global__ void __launch_bounds__(1024)
lstm_scan_kernel(const float* __restrict__ xz,
                 const unsigned short* __restrict__ wfrag,
                 const float* __restrict__ h0,
                 const float* __restrict__ c0,
                 float* __restrict__ out) {
  extern __shared__ char smem[];
  unsigned short* hbuf = (unsigned short*)smem;           // [2][112][224] bf16
  float* cbuf  = (float*)(smem + CBUF_OFF);               // [112][200] f32
  float* stage = (float*)(smem + STAGE_OFF);              // [32][256] f32

  const int tid  = threadIdx.x;
  const int lane = tid & 31;
  const int wave = tid >> 5;
  const int lhi  = (lane & 16) ? 1 : 0;
  const int l15  = lane & 15;

  // init: both h buffers = h0 (bf16, zero-padded rows/cols), c = c0
  for (int idx = tid; idx < 2 * 112 * KPAD_REC; idx += 1024) {
    int r = idx % (112 * KPAD_REC);
    int m = r / KPAD_REC, k = r % KPAD_REC;
    unsigned short v = 0;
    if (m < B_ && k < U_) v = f2bf(h0[m * U_ + k]);
    hbuf[idx] = v;
  }
  for (int idx = tid; idx < 112 * U_; idx += 1024) {
    int m = idx / U_, j = idx % U_;
    cbuf[idx] = (m < B_) ? c0[m * U_ + j] : 0.0f;
  }
  __syncthreads();

  for (int t = 0; t < T_; ++t) {
    const int cur = t & 1;
    const unsigned short* hb = hbuf + cur * (112 * KPAD_REC);
    unsigned short* hw = hbuf + (cur ^ 1) * (112 * KPAD_REC);
    const int tn = (t + 1 < T_) ? t + 1 : t;   // next step for prefetch

    for (int tile = wave; tile < 7 * NT; tile += 32) {   // static tile->wave map
      const int mt = tile / NT;
      const int nt = tile % NT;
      const int col = nt * 16 + l15;

      // acc := xz tile (C-fragment layout), clamped rows for padding;
      // prefetch next timestep's rows to overlap HBM latency with WMMA.
      v8f acc;
#pragma unroll
      for (int r = 0; r < 8; ++r) {
        int m = mt * 16 + r + lhi * 8;
        int mc = m < B_ ? m : B_ - 1;
        const float* src = xz + ((size_t)mc * T_ + t) * FOURU + col;
        acc[r] = *src;
        __builtin_prefetch(xz + ((size_t)mc * T_ + tn) * FOURU + col, 0, 1);
      }

      const unsigned short* arow = hb + (mt * 16 + l15) * KPAD_REC;
      const int kb0 = lhi * 16;
#pragma unroll
      for (int kc = 0; kc < 7; ++kc) {
        const int kb = kb0 + kc * 32;
        U32x8 at, bt;
        at.lo = *(const uint4*)(arow + kb);        // ds_load_b128
        at.hi = *(const uint4*)(arow + kb + 8); 
        const unsigned short* bp = wfrag + (((size_t)nt * 7 + kc) * 32 + lane) * 16;
        bt.lo = *(const uint4*)(bp);               // L2-resident fragments
        bt.hi = *(const uint4*)(bp + 8);
        v16bf av = __builtin_bit_cast(v16bf, at);
        v16bf bv = __builtin_bit_cast(v16bf, bt);
        acc = __builtin_amdgcn_wmma_f32_16x16x32_bf16(false, av, false, bv,
                                                      (short)0, acc, false, false);
      }

      // stage z tile (wave-private LDS, no barrier needed within a wave)
      float* st = stage + wave * 256;
#pragma unroll
      for (int r = 0; r < 8; ++r)
        st[(r + lhi * 8) * 16 + l15] = acc[r];

      if (mt < 6)  gate_update<false>(st, cbuf, hw, out, mt, nt, t, lane);
      else         gate_update<true >(st, cbuf, hw, out, mt, nt, t, lane);
    }
    __syncthreads();   // h double-buffer handoff
  }
}

// ---------------------------------------------------------------------------
static inline size_t alignup(size_t x) { return (x + 255) & ~(size_t)255; }

extern "C" void kernel_launch(void* const* d_in, const int* in_sizes, int n_in,
                              void* d_out, int out_size, void* d_ws, size_t ws_size,
                              hipStream_t stream) {
  (void)in_sizes; (void)n_in; (void)out_size; (void)ws_size;
  const int*   tokens     = (const int*)d_in[0];
  const float* emb_table  = (const float*)d_in[1];
  const float* in_kernel  = (const float*)d_in[2];
  const float* rec_kernel = (const float*)d_in[3];
  const float* bias       = (const float*)d_in[4];
  const float* h0         = (const float*)d_in[5];
  const float* c0         = (const float*)d_in[6];
  float* out = (float*)d_out;

  // workspace carve
  char* p = (char*)d_ws;
  float* xz = (float*)p;                 p += alignup((size_t)MROWS * FOURU * 4);
  unsigned short* embbf = (unsigned short*)p; p += alignup((size_t)V_ * KPAD_IN * 2);
  float* biasp = (float*)p;              p += alignup((size_t)FOURU * 4);
  unsigned short* kfrag = (unsigned short*)p; p += alignup((size_t)NT * 4 * 32 * 16 * 2);
  unsigned short* wfrag = (unsigned short*)p; p += alignup((size_t)NT * 7 * 32 * 16 * 2);

  prep_kernel<<<512, 256, 0, stream>>>(emb_table, in_kernel, rec_kernel, bias,
                                       embbf, biasp, kfrag, wfrag);
  input_gemm_kernel<<<2048, 256, 0, stream>>>(tokens, embbf, kfrag, biasp, xz);

  (void)hipFuncSetAttribute((const void*)lstm_scan_kernel,
                            hipFuncAttributeMaxDynamicSharedMemorySize,
                            SMEM_BYTES);
  lstm_scan_kernel<<<1, 1024, SMEM_BYTES, stream>>>(xz, wfrag, h0, c0, out);
}
